// IVA_Module_88210038325549
// MI455X (gfx1250) — matrix operand order repaired and checked by reading, compiled, not run on hardware
//
#include <hip/hip_runtime.h>

// ---------------------------------------------------------------------------
// IVA cross-attention for MI455X (gfx1250, wave32, WMMA + TDM).
//   out1[b,c,m] = sum_n softmax_n(Q[b,m,:]·K[b,n,:]) * Vimg[b,c,n]
//   out2 likewise with Vpc.
// Phase 1: bf16 projections (Q,K row-major [N,256]; Vimg,Vpc [256,N]).
// Phase 2: flash attention, V_WMMA_F32_16X16X32_BF16, online softmax,
//          K tiles staged to LDS via TENSOR_LOAD_TO_LDS (TDM) when available.
// ---------------------------------------------------------------------------

typedef __attribute__((ext_vector_type(16))) __bf16 v16bf;
typedef __attribute__((ext_vector_type(8)))  __bf16 v8bf;
typedef __attribute__((ext_vector_type(4)))  __bf16 v4bf;
typedef __attribute__((ext_vector_type(8)))  float  v8f;
typedef __attribute__((ext_vector_type(4)))  unsigned int u32x4;
typedef __attribute__((ext_vector_type(8)))  int i32x8;
typedef __attribute__((ext_vector_type(4)))  int i32x4;

#define N_PIX 4096
#define CDIM  256
#define LDK   272   // padded LDS stride (bf16) for [pixel][ch] / K tiles
#define SLD   68    // padded LDS stride (f32) for the 32x64 score tile

#if defined(__gfx1250__) && __has_builtin(__builtin_amdgcn_tensor_load_to_lds)
#define IVA_TDM 1
#else
#define IVA_TDM 0
#endif

static __device__ __forceinline__ v8f v8f_zero() {
  v8f z = {0.f, 0.f, 0.f, 0.f, 0.f, 0.f, 0.f, 0.f};
  return z;
}

static __device__ __forceinline__ v8f wmma_bf16(v16bf a, v16bf b, v8f c) {
  // D = A(16x32 bf16) * B(32x16 bf16) + C(16x16 f32)
  return __builtin_amdgcn_wmma_f32_16x16x32_bf16(
      false, a, false, b, (short)0, c, false, false);
}

// A fragment (16x32) from a row-major bf16 row: two contiguous 16B chunks.
// lanes 0-15 hold K {k..k+15}, lanes 16-31 hold {k+16..k+31}.
static __device__ __forceinline__ v16bf load_a_row_bf16(
    const __bf16* __restrict__ row, int k, int lane) {
  const int g = (lane >> 4) * 16;
  const v8bf lo = *(const v8bf*)(row + k + g);
  const v8bf hi = *(const v8bf*)(row + k + g + 8);
  return __builtin_shufflevector(lo, hi, 0, 1, 2, 3, 4, 5, 6, 7,
                                 8, 9, 10, 11, 12, 13, 14, 15);
}

// A fragment from f32 weights (cvt to bf16 in regs).
static __device__ __forceinline__ v16bf load_a_row_f32(
    const float* __restrict__ row, int k, int lane) {
  const int g = (lane >> 4) * 16;
  const float4 x0 = *(const float4*)(row + k + g);
  const float4 x1 = *(const float4*)(row + k + g + 4);
  const float4 x2 = *(const float4*)(row + k + g + 8);
  const float4 x3 = *(const float4*)(row + k + g + 12);
  v16bf a;
  a[0] = (__bf16)x0.x;  a[1] = (__bf16)x0.y;
  a[2] = (__bf16)x0.z;  a[3] = (__bf16)x0.w;
  a[4] = (__bf16)x1.x;  a[5] = (__bf16)x1.y;
  a[6] = (__bf16)x1.z;  a[7] = (__bf16)x1.w;
  a[8]  = (__bf16)x2.x; a[9]  = (__bf16)x2.y;
  a[10] = (__bf16)x2.z; a[11] = (__bf16)x2.w;
  a[12] = (__bf16)x3.x; a[13] = (__bf16)x3.y;
  a[14] = (__bf16)x3.z; a[15] = (__bf16)x3.w;
  return a;
}

// ---------------------------------------------------------------------------
// Kernel 1: projections.  grid = (64 pixel-tiles, B, 2 sources), block = 256.
// LDS X tile is [pixel][in-ch] (stride LDK) so WMMA B-fragments are a single
// contiguous 32B load per lane.
// ---------------------------------------------------------------------------
__global__ __launch_bounds__(256) void iva_proj_kernel(
    const float* __restrict__ img, const float* __restrict__ pc,
    const float* __restrict__ qw,  const float* __restrict__ qb,
    const float* __restrict__ kw,  const float* __restrict__ kb,
    const float* __restrict__ viw, const float* __restrict__ vib,
    const float* __restrict__ vpw, const float* __restrict__ vpb,
    __bf16* __restrict__ Qws, __bf16* __restrict__ Kws,
    __bf16* __restrict__ Viws, __bf16* __restrict__ Vpws) {
  __shared__ __bf16 Xs[64 * LDK];  // [pixel][in-ch], 34 KB

  const int n0  = blockIdx.x * 64;
  const int b   = blockIdx.y;
  const int src = blockIdx.z;
  const int t   = threadIdx.x;

  // ---- stage input tile into LDS (bf16, transposed to [pixel][ch]) ----
  if (src == 0) {
    // img is [B, 256, 4096]: coalesced loads along n, packed v4bf LDS stores
    const int n = t & 63, cb = t >> 6;
    #pragma unroll
    for (int cc = 0; cc < 16; ++cc) {
      const int c4 = cb * 64 + cc * 4;
      const float* s = img + ((size_t)b * CDIM + c4) * N_PIX + n0 + n;
      v4bf pk;
      pk[0] = (__bf16)s[0];
      pk[1] = (__bf16)s[(size_t)N_PIX];
      pk[2] = (__bf16)s[(size_t)2 * N_PIX];
      pk[3] = (__bf16)s[(size_t)3 * N_PIX];
      *(v4bf*)&Xs[n * LDK + c4] = pk;
    }
  } else {
    // pc is [B, 16(D), 4096, 16(C)]; pc2 channel j = c*16 + d.
    // Thread (c,n) gathers d=0..15 -> 16 consecutive j -> two v8bf stores.
    #pragma unroll
    for (int it = 0; it < 4; ++it) {
      const int idx = t + 256 * it;
      const int c = idx >> 6, n = idx & 63;
      const float* s = pc + ((size_t)b * 16 * N_PIX + n0 + n) * 16 + c;
      v8bf lo, hi;
      #pragma unroll
      for (int d = 0; d < 8; ++d) {
        lo[d] = (__bf16)s[(size_t)d * (N_PIX * 16)];
        hi[d] = (__bf16)s[(size_t)(d + 8) * (N_PIX * 16)];
      }
      *(v8bf*)&Xs[n * LDK + c * 16]     = lo;
      *(v8bf*)&Xs[n * LDK + c * 16 + 8] = hi;
    }
  }
  __syncthreads();

  const int w = t >> 5, lane = t & 31;
  const int o0 = w * 32;  // 32 output channels per wave

  for (int pass = 0; pass < 2; ++pass) {
    const float* W    = (src == 0) ? (pass ? viw : qw) : (pass ? vpw : kw);
    const float* bias = (src == 0) ? (pass ? vib : qb) : (pass ? vpb : kb);

    v8f acc[2][4];
    #pragma unroll
    for (int i = 0; i < 2; ++i)
      #pragma unroll
      for (int j = 0; j < 4; ++j) acc[i][j] = v8f_zero();

    #pragma unroll
    for (int k = 0; k < CDIM; k += 32) {
      const v16bf a0 =
          load_a_row_f32(W + (size_t)(o0 + (lane & 15)) * CDIM, k, lane);
      const v16bf a1 =
          load_a_row_f32(W + (size_t)(o0 + 16 + (lane & 15)) * CDIM, k, lane);
      #pragma unroll
      for (int nt = 0; nt < 4; ++nt) {
        // B fragment: 16 contiguous bf16 per lane (32B aligned)
        const v16bf bb = *(const v16bf*)&Xs[(nt * 16 + (lane & 15)) * LDK +
                                            k + ((lane >> 4) << 4)];
        acc[0][nt] = wmma_bf16(a0, bb, acc[0][nt]);
        acc[1][nt] = wmma_bf16(a1, bb, acc[1][nt]);
      }
    }

    if (pass == 0) {
      // Q / K: row-major [N][256] bf16, one v8bf (16B) store per tile
      __bf16* dst = ((src == 0) ? Qws : Kws) + (size_t)b * N_PIX * CDIM;
      #pragma unroll
      for (int ot = 0; ot < 2; ++ot) {
        const int o = o0 + ot * 16 + ((lane >> 4) << 3);
        const float4 b0 = *(const float4*)(bias + o);
        const float4 b1 = *(const float4*)(bias + o + 4);
        #pragma unroll
        for (int nt = 0; nt < 4; ++nt) {
          const int n = n0 + nt * 16 + (lane & 15);
          v8bf pk;
          pk[0] = (__bf16)(acc[ot][nt][0] + b0.x);
          pk[1] = (__bf16)(acc[ot][nt][1] + b0.y);
          pk[2] = (__bf16)(acc[ot][nt][2] + b0.z);
          pk[3] = (__bf16)(acc[ot][nt][3] + b0.w);
          pk[4] = (__bf16)(acc[ot][nt][4] + b1.x);
          pk[5] = (__bf16)(acc[ot][nt][5] + b1.y);
          pk[6] = (__bf16)(acc[ot][nt][6] + b1.z);
          pk[7] = (__bf16)(acc[ot][nt][7] + b1.w);
          *(v8bf*)&dst[(size_t)n * CDIM + o] = pk;
        }
      }
    } else {
      // Vimg / Vpc: channel-major [256][N] bf16 (lanes coalesce along n)
      __bf16* dst = ((src == 0) ? Viws : Vpws) + (size_t)b * N_PIX * CDIM;
      #pragma unroll
      for (int ot = 0; ot < 2; ++ot) {
        const int o = o0 + ot * 16 + ((lane >> 4) << 3);
        const float4 b0 = *(const float4*)(bias + o);
        const float4 b1 = *(const float4*)(bias + o + 4);
        const float bb8[8] = {b0.x, b0.y, b0.z, b0.w, b1.x, b1.y, b1.z, b1.w};
        #pragma unroll
        for (int nt = 0; nt < 4; ++nt) {
          const int n = n0 + nt * 16 + (lane & 15);
          #pragma unroll
          for (int v = 0; v < 8; ++v)
            dst[(size_t)(o + v) * N_PIX + n] =
                (__bf16)(acc[ot][nt][v] + bb8[v]);
        }
      }
    }
  }
}

// ---------------------------------------------------------------------------
// Kernel 2: flash attention.  grid = (128 row-tiles, B), block = 256 (8 waves)
// Wave w: mg = w>>2 -> 16-row slice, cg = w&3 -> 128 output channels from
// {Vimg lo/hi, Vpc lo/hi}.  KV tile = 64.  K tile staged to LDS by the TDM
// (1-D 32KB descriptor with pad_enable so rows land at stride LDK bf16).
// ---------------------------------------------------------------------------
__global__ __launch_bounds__(256) void iva_attn_kernel(
    const __bf16* __restrict__ Qws, const __bf16* __restrict__ Kws,
    const __bf16* __restrict__ Viws, const __bf16* __restrict__ Vpws,
    float* __restrict__ out) {
  __shared__ __bf16 Ks[64 * LDK];  // K tile, padded rows (34 KB)
  __shared__ float  S[32 * SLD];   // score block (8.5 KB)

  const int m0 = blockIdx.x * 32;
  const int b  = blockIdx.y;
  const int t = threadIdx.x, w = t >> 5, lane = t & 31;
  const int mg = w >> 2, cg = w & 3;

  const __bf16* Q = Qws + (size_t)b * N_PIX * CDIM;
  const __bf16* K = Kws + (size_t)b * N_PIX * CDIM;
  const __bf16* V = ((cg < 2) ? Viws : Vpws) + (size_t)b * N_PIX * CDIM;
  const int c0 = (cg & 1) * 128;

  float m_run = -1e30f, l_run = 0.f;
  v8f acc[8];  // 16 rows x 128 channels of O per wave
  #pragma unroll
  for (int i = 0; i < 8; ++i) acc[i] = v8f_zero();

  // Q A-fragments are loop-invariant: preload all 8 (256 channels).
  const __bf16* qrow = Q + (size_t)(m0 + mg * 16 + (lane & 15)) * CDIM;
  v16bf qfrag[8];
  #pragma unroll
  for (int kc = 0; kc < 8; ++kc)
    qfrag[kc] = load_a_row_bf16(qrow, kc * 32, lane);

  for (int kv0 = 0; kv0 < N_PIX; kv0 += 64) {
    __syncthreads();  // previous tile's Ks / S consumers done

    // ---- stage K tile [64 x 256] bf16 -> LDS rows at stride LDK ----
#if IVA_TDM
    if (w == 0) {
      const unsigned lds = (unsigned)(size_t)(void*)Ks;
      const unsigned long long ga =
          (unsigned long long)(size_t)(const void*)(K + (size_t)kv0 * CDIM);
      u32x4 g0;
      g0[0] = 1u;                                   // count=1 (user D#)
      g0[1] = lds;                                  // lds_addr
      g0[2] = (unsigned)(ga & 0xffffffffull);       // global_addr[31:0]
      g0[3] = (unsigned)((ga >> 32) & 0x01ffffffu)  // global_addr[56:32]
              | (2u << 30);                         // type=2 (image)
      i32x8 g1;
      g1[0] = (1 << 16)     // data_size = 2 bytes
            | (1 << 20)     // pad_enable (LDS-side padding)
            | (6 << 22)     // pad_interval code 6 -> every 128 dwords
            | (7 << 25);    // pad_amount  code 7 -> 8 dwords (16 bf16)
      g1[1] = (int)(16384u << 16);   // tensor_dim0[15:0] = 16384 elements
      g1[2] = (int)(1u << 16);       // tensor_dim0 hi = 0, tensor_dim1 = 1
      g1[3] = (int)(16384u << 16);   // tile_dim0 = 16384 (1-D 32KB tile)
      g1[4] = 0;                     // tile_dim1 = tile_dim2 = 0 (unused)
      g1[5] = 16384;                 // tensor_dim0_stride lo
      g1[6] = 0;
      g1[7] = 0;
      i32x4 g2 = {0, 0, 0, 0};
      i32x4 g3 = {0, 0, 0, 0};
      i32x8 g4 = {0, 0, 0, 0, 0, 0, 0, 0};  // extra group (6-arg builtin)
      __builtin_amdgcn_tensor_load_to_lds(g0, g1, g2, g3, g4, 0);
      __builtin_amdgcn_s_wait_tensorcnt(0);
    }
#else
    // cooperative fallback copy (also used by the host-side pass)
    #pragma unroll
    for (int i = 0; i < 4; ++i) {
      const int idx = t + 256 * i;
      const int r = idx >> 4, ck = idx & 15;
      *(v16bf*)&Ks[r * LDK + ck * 16] =
          *(const v16bf*)&K[(size_t)(kv0 + r) * CDIM + ck * 16];
    }
#endif
    __builtin_prefetch(V + (size_t)(c0 + mg * 64 + (lane & 15)) * N_PIX +
                           kv0 + 64, 0, 1);
    __syncthreads();  // K tile visible to all waves

    // ---- S subtile: rows mg*16.., cols kv0 + cg*16.. (16x16) ----
    v8f s = v8f_zero();
    #pragma unroll
    for (int kc = 0; kc < 8; ++kc) {
      const v16bf bb = *(const v16bf*)&Ks[(cg * 16 + (lane & 15)) * LDK +
                                          kc * 32 + ((lane >> 4) << 4)];
      s = wmma_bf16(qfrag[kc], bb, s);
    }
    {
      const int r  = mg * 16 + ((lane >> 4) << 3);
      const int cS = cg * 16 + (lane & 15);
      #pragma unroll
      for (int v = 0; v < 8; ++v) S[(r + v) * SLD + cS] = s[v];
    }
    __syncthreads();  // full 32x64 score block visible

    // ---- online softmax stats: lane r owns row r (0..31) ----
    const float* srow = &S[lane * SLD];
    float tmax = -1e30f;
    for (int c = 0; c < 64; ++c) tmax = fmaxf(tmax, srow[c]);
    const float m_new = fmaxf(m_run, tmax);
    const float alpha = __expf(m_run - m_new);
    float psum = 0.f;
    for (int c = 0; c < 64; ++c) psum += __expf(srow[c] - m_new);
    l_run = l_run * alpha + psum;
    m_run = m_new;

    // rescale O accumulators by this tile's alpha (per accumulator row)
    #pragma unroll
    for (int v = 0; v < 8; ++v) {
      const float av = __shfl(alpha, mg * 16 + ((lane >> 4) << 3) + v, 32);
      #pragma unroll
      for (int ct = 0; ct < 8; ++ct) acc[ct][v] *= av;
    }

    // ---- O += P @ V  (P from LDS with exp applied, A layout) ----
    const float mrow = __shfl(m_new, mg * 16 + (lane & 15), 32);
    const int rbase = (mg * 16 + (lane & 15)) * SLD;
    #pragma unroll
    for (int kkh = 0; kkh < 2; ++kkh) {
      const int kk = kkh * 32;
      const int g = (lane >> 4) * 16;
      const float4 p0 = *(const float4*)&S[rbase + kk + g];
      const float4 p1 = *(const float4*)&S[rbase + kk + g + 4];
      const float4 p2 = *(const float4*)&S[rbase + kk + g + 8];
      const float4 p3 = *(const float4*)&S[rbase + kk + g + 12];
      v16bf pa;
      pa[0]  = (__bf16)__expf(p0.x - mrow);
      pa[1]  = (__bf16)__expf(p0.y - mrow);
      pa[2]  = (__bf16)__expf(p0.z - mrow);
      pa[3]  = (__bf16)__expf(p0.w - mrow);
      pa[4]  = (__bf16)__expf(p1.x - mrow);
      pa[5]  = (__bf16)__expf(p1.y - mrow);
      pa[6]  = (__bf16)__expf(p1.z - mrow);
      pa[7]  = (__bf16)__expf(p1.w - mrow);
      pa[8]  = (__bf16)__expf(p2.x - mrow);
      pa[9]  = (__bf16)__expf(p2.y - mrow);
      pa[10] = (__bf16)__expf(p2.z - mrow);
      pa[11] = (__bf16)__expf(p2.w - mrow);
      pa[12] = (__bf16)__expf(p3.x - mrow);
      pa[13] = (__bf16)__expf(p3.y - mrow);
      pa[14] = (__bf16)__expf(p3.z - mrow);
      pa[15] = (__bf16)__expf(p3.w - mrow);
      #pragma unroll
      for (int ct = 0; ct < 8; ++ct) {
        const v16bf vb = *(const v16bf*)(V +
            (size_t)(c0 + ct * 16 + (lane & 15)) * N_PIX +
            kv0 + kk + ((lane >> 4) << 4));
        acc[ct] = wmma_bf16(pa, vb, acc[ct]);
      }
    }
  }

  // ---- epilogue: divide by l, write [B,256,N] (img then pc block) ----
  float* obase = out + ((cg < 2) ? (size_t)0 : (size_t)2 * CDIM * N_PIX);
  float linv[8];
  #pragma unroll
  for (int v = 0; v < 8; ++v) {
    const float l = __shfl(l_run, mg * 16 + ((lane >> 4) << 3) + v, 32);
    linv[v] = 1.f / l;
  }
  #pragma unroll
  for (int ct = 0; ct < 8; ++ct) {
    const int ch = c0 + ct * 16 + (lane & 15);
    const int m  = m0 + mg * 16 + ((lane >> 4) << 3);
    float* dst = obase + ((size_t)b * CDIM + ch) * N_PIX + m;
    #pragma unroll
    for (int v = 0; v < 8; ++v) dst[v] = acc[ct][v] * linv[v];
  }
}

// ---------------------------------------------------------------------------
extern "C" void kernel_launch(void* const* d_in, const int* in_sizes, int n_in,
                              void* d_out, int out_size, void* d_ws,
                              size_t ws_size, hipStream_t stream) {
  const float* img = (const float*)d_in[0];
  const float* pc  = (const float*)d_in[1];
  const float* qw  = (const float*)d_in[2];
  const float* qb  = (const float*)d_in[3];
  const float* kw  = (const float*)d_in[4];
  const float* kb  = (const float*)d_in[5];
  const float* viw = (const float*)d_in[6];
  const float* vib = (const float*)d_in[7];
  const float* vpw = (const float*)d_in[8];
  const float* vpb = (const float*)d_in[9];

  // workspace: 4 bf16 arrays of B*N*C = 2*4096*256 elements (4 MB each)
  const size_t NE = (size_t)2 * N_PIX * CDIM;
  __bf16* Qws  = (__bf16*)d_ws;
  __bf16* Kws  = Qws + NE;
  __bf16* Viws = Kws + NE;
  __bf16* Vpws = Viws + NE;

  iva_proj_kernel<<<dim3(64, 2, 2), 256, 0, stream>>>(
      img, pc, qw, qb, kw, kb, viw, vib, vpw, vpb, Qws, Kws, Viws, Vpws);

  iva_attn_kernel<<<dim3(128, 2), 256, 0, stream>>>(
      Qws, Kws, Viws, Vpws, (float*)d_out);
}
